// GroupWiseMoE_58299886076202
// MI455X (gfx1250) — compile-verified, matrix operand order checked by reading, not run on hardware
//
#include <hip/hip_runtime.h>
#include <hip/hip_bf16.h>

// ---------------- problem constants ----------------
#define NTOK 8192
#define DIN  1024
#define HID  2048
#define NEXP 8
#define TOPK 2
#define NPAIR (NTOK * TOPK)      // 16384 (token,expert) pairs
#define CAP   NPAIR              // worst-case pairs per expert

// ---------------- GEMM tiling ----------------
#define BM   128                 // pairs per tile
#define BN   128                 // output cols per tile
#define BKT  64                  // K per LDS tile (2 WMMA k-steps)
#define APAD 72                  // padded row length in LDS (bf16 elems)
#define TILES_M (NPAIR / BM)     // 128

typedef __attribute__((ext_vector_type(16))) __bf16 v16bf;
typedef __attribute__((ext_vector_type(8)))  __bf16 v8bf;
typedef __attribute__((ext_vector_type(4)))  __bf16 v4bf;
typedef __attribute__((ext_vector_type(8)))  float  v8f;

// gfx1250 async copy: LDS[lds_off] <= MEM[gaddr], 16B per lane, ASYNCcnt.
__device__ __forceinline__ void async_copy_b128(unsigned int lds_off,
                                                unsigned long long gaddr) {
    asm volatile("global_load_async_to_lds_b128 %0, %1, off"
                 :: "v"(lds_off), "v"(gaddr) : "memory");
}
__device__ __forceinline__ unsigned int lds_addr(const void* p) {
    return (unsigned int)(unsigned long long)p;   // low 32 bits = LDS offset
}

// ---------------------------------------------------------------------------
// Kernel 0: zero output accumulator + expert counters.
// ---------------------------------------------------------------------------
__global__ __launch_bounds__(256) void moe_init_kernel(float* __restrict__ out,
                                                       int* __restrict__ counts) {
    size_t total = (size_t)NTOK * HID;
    for (size_t i = (size_t)blockIdx.x * blockDim.x + threadIdx.x; i < total;
         i += (size_t)gridDim.x * blockDim.x)
        out[i] = 0.0f;
    if (blockIdx.x == 0 && threadIdx.x < NEXP) counts[threadIdx.x] = 0;
}

// ---------------------------------------------------------------------------
// Kernel 1: router. One wave32 per token.
// ---------------------------------------------------------------------------
__global__ __launch_bounds__(256) void moe_router_kernel(
    const float* __restrict__ x, const float* __restrict__ Wg,
    const float* __restrict__ bg,
    float* __restrict__ probs_out, float* __restrict__ mask_out,
    int* __restrict__ counts, int* __restrict__ pairs,
    float* __restrict__ pprob) {
    int wave = threadIdx.x >> 5;
    int lane = threadIdx.x & 31;
    int n = blockIdx.x * 8 + wave;
    if (n >= NTOK) return;

    float acc[NEXP];
#pragma unroll
    for (int e = 0; e < NEXP; e++) acc[e] = 0.0f;

    const float* xr = x + (size_t)n * DIN;
    for (int d = lane; d < DIN; d += 32) {
        float xv = xr[d];
        const float* wr = Wg + (size_t)d * NEXP;
#pragma unroll
        for (int e = 0; e < NEXP; e++) acc[e] += xv * wr[e];
    }
#pragma unroll
    for (int off = 16; off > 0; off >>= 1) {
#pragma unroll
        for (int e = 0; e < NEXP; e++) acc[e] += __shfl_xor(acc[e], off, 32);
    }

    if (lane == 0) {
        float logits[NEXP], mx = -1e30f;
#pragma unroll
        for (int e = 0; e < NEXP; e++) {
            logits[e] = acc[e] + bg[e];
            mx = fmaxf(mx, logits[e]);
        }
        float p[NEXP], s = 0.0f;
#pragma unroll
        for (int e = 0; e < NEXP; e++) { p[e] = __expf(logits[e] - mx); s += p[e]; }
        float inv = 1.0f / s;
#pragma unroll
        for (int e = 0; e < NEXP; e++) {
            p[e] *= inv;
            probs_out[(size_t)n * NEXP + e] = p[e];
        }
        int e0 = 0;
#pragma unroll
        for (int e = 1; e < NEXP; e++) if (p[e] > p[e0]) e0 = e;
        int e1 = -1;
#pragma unroll
        for (int e = 0; e < NEXP; e++)
            if (e != e0 && (e1 < 0 || p[e] > p[e1])) e1 = e;

        float den = p[e0] + p[e1] + 1e-8f;
        float q0 = p[e0] / den, q1 = p[e1] / den;
#pragma unroll
        for (int e = 0; e < NEXP; e++)
            mask_out[(size_t)n * NEXP + e] = (e == e0) ? q0 : ((e == e1) ? q1 : 0.0f);

        int s0 = atomicAdd(&counts[e0], 1);
        pairs[e0 * CAP + s0] = n * 2 + 0;
        int s1 = atomicAdd(&counts[e1], 1);
        pairs[e1 * CAP + s1] = n * 2 + 1;
        pprob[n * 2 + 0] = q0;
        pprob[n * 2 + 1] = q1;
    }
}

// ---------------------------------------------------------------------------
// Kernel 2: convert x -> bf16 (contiguous).
// ---------------------------------------------------------------------------
__global__ __launch_bounds__(256) void moe_cvtx_kernel(const float* __restrict__ x,
                                                       __bf16* __restrict__ xb) {
    size_t quads = (size_t)NTOK * DIN / 4;
    for (size_t i = (size_t)blockIdx.x * blockDim.x + threadIdx.x; i < quads;
         i += (size_t)gridDim.x * blockDim.x) {
        const float4 v = ((const float4*)x)[i];
        v4bf o;
        o[0] = (__bf16)v.x; o[1] = (__bf16)v.y;
        o[2] = (__bf16)v.z; o[3] = (__bf16)v.w;
        *(v4bf*)(xb + 4 * i) = o;
    }
}

// ---------------------------------------------------------------------------
// Kernel 3: per-expert transpose + convert: in [E][R][C] f32 -> out [E][C][R] bf16
// ---------------------------------------------------------------------------
__global__ __launch_bounds__(256) void moe_transpose_kernel(
    const float* __restrict__ in, __bf16* __restrict__ out, int R, int C) {
    __shared__ __bf16 t[32][33];
    int e = blockIdx.z;
    const float* ip = in + (size_t)e * R * C;
    __bf16* op = out + (size_t)e * R * C;
    int c0 = blockIdx.x * 32, r0 = blockIdx.y * 32;
    int tx = threadIdx.x & 31, ty = threadIdx.x >> 5;   // 32 x 8
#pragma unroll
    for (int j = 0; j < 4; j++)
        t[ty + j * 8][tx] = (__bf16)ip[(size_t)(r0 + ty + j * 8) * C + c0 + tx];
    __syncthreads();
#pragma unroll
    for (int j = 0; j < 4; j++)
        op[(size_t)(c0 + ty + j * 8) * R + r0 + tx] = t[tx][ty + j * 8];
}

// ---------------------------------------------------------------------------
// Fused GEMM kernel (layer 1 or layer 2), bf16 WMMA, double-buffered LDS fed
// by global_load_async_to_lds_b128.
//   A   : [*, KD] bf16   (xb gathered by token, or hbuf gathered by pair)
//   BT  : [E][HID][KD] bf16  (transposed weights: row = out col, len KD)
//   IS2 : false -> h = relu(A W1 + b1) -> hbuf (bf16)
//         true  -> out += pprob * (h W2 + b2)  (atomic f32)
// Block: 256 threads = 8 waves as 4(M) x 2(N); wave computes 32x64 via 8 frags.
// ---------------------------------------------------------------------------
template <int KD, bool IS2>
__global__ __launch_bounds__(256) void moe_ffn_kernel(
    const __bf16* __restrict__ A, const __bf16* __restrict__ BT,
    const float* __restrict__ bias,
    const int* __restrict__ counts, const int* __restrict__ pairs,
    const float* __restrict__ pprob,
    __bf16* __restrict__ hbuf, float* __restrict__ out) {
    int e    = blockIdx.x / TILES_M;
    int tm   = blockIdx.x % TILES_M;
    int cnt  = counts[e];
    int base = tm * BM;
    if (base >= cnt) return;
    int n0 = blockIdx.y * BN;

    __shared__ __align__(16) __bf16 As[2][BM][APAD];
    __shared__ __align__(16) __bf16 Bs[2][BN][APAD];
    __shared__ int rowPair[BM];

    int tid = threadIdx.x;
    if (tid < BM) {
        int idx = base + tid;
        rowPair[tid] = (idx < cnt) ? pairs[e * CAP + idx] : -1;
    }
    __syncthreads();

    // ---- precompute per-thread async copy slots (4 A chunks + 4 B chunks) ----
    const __bf16* Bbase = BT + (size_t)e * HID * KD + (size_t)n0 * KD;
    unsigned long long aSrc[4], bSrc[4];
    unsigned int aDst[4], bDst[4];
#pragma unroll
    for (int u = 0; u < 4; u++) {
        int lin = tid + u * 256;          // 16B-chunk id, 0..1023
        int r = lin >> 3, ch = lin & 7;   // row, chunk-in-row (8 x 16B = 128B)
        int p = rowPair[r];
        int src = (p < 0) ? 0 : p;        // clamp: pair 0 always valid
        if (!IS2) src >>= 1;              // token id for layer 1
        aSrc[u] = (unsigned long long)(A + (size_t)src * KD + ch * 8);
        aDst[u] = lds_addr(&As[0][r][ch * 8]);
        bSrc[u] = (unsigned long long)(Bbase + (size_t)r * KD + ch * 8);
        bDst[u] = lds_addr(&Bs[0][r][ch * 8]);
    }
    const unsigned int aBufStride = lds_addr(&As[1][0][0]) - lds_addr(&As[0][0][0]);
    const unsigned int bBufStride = lds_addr(&Bs[1][0][0]) - lds_addr(&Bs[0][0][0]);

    int wv = tid >> 5, lane = tid & 31;
    int wm = wv & 3, wn = wv >> 2;
    int lm = lane & 15;
    int koff = (lane >> 4) * 8;

    v8f acc[2][4] = {};

    const int T = KD / BKT;
    // prologue: tile 0 -> buffer 0   (8 async instrs per wave)
#pragma unroll
    for (int u = 0; u < 4; u++) {
        async_copy_b128(aDst[u], aSrc[u]);
        async_copy_b128(bDst[u], bSrc[u]);
    }

    for (int t = 0; t < T; t++) {
        if (t + 1 < T) {
            int nb = (t + 1) & 1;
            size_t kb = (size_t)(t + 1) * BKT * sizeof(__bf16);
#pragma unroll
            for (int u = 0; u < 4; u++) {
                async_copy_b128(aDst[u] + nb * aBufStride, aSrc[u] + kb);
                async_copy_b128(bDst[u] + nb * bBufStride, bSrc[u] + kb);
            }
            asm volatile("s_wait_asynccnt 0x8" ::: "memory");  // tile t landed
        } else {
            asm volatile("s_wait_asynccnt 0x0" ::: "memory");
        }
        __syncthreads();   // tile t visible to all waves

        int cb = t & 1;
#pragma unroll
        for (int s = 0; s < 2; s++) {     // two k=32 substeps
            int kb = s * 32;
            v16bf afr[2], bfr[4];
#pragma unroll
            for (int sm = 0; sm < 2; sm++) {
                int row = wm * 32 + sm * 16 + lm;
                v8bf lo = *(const v8bf*)&As[cb][row][kb + koff];
                v8bf hi = *(const v8bf*)&As[cb][row][kb + koff + 16];
                afr[sm] = __builtin_shufflevector(lo, hi, 0, 1, 2, 3, 4, 5, 6, 7,
                                                  8, 9, 10, 11, 12, 13, 14, 15);
            }
#pragma unroll
            for (int sn = 0; sn < 4; sn++) {
                int col = wn * 64 + sn * 16 + lm;
                v8bf lo = *(const v8bf*)&Bs[cb][col][kb + koff];
                v8bf hi = *(const v8bf*)&Bs[cb][col][kb + koff + 16];
                bfr[sn] = __builtin_shufflevector(lo, hi, 0, 1, 2, 3, 4, 5, 6, 7,
                                                  8, 9, 10, 11, 12, 13, 14, 15);
            }
#pragma unroll
            for (int sm = 0; sm < 2; sm++)
#pragma unroll
                for (int sn = 0; sn < 4; sn++)
                    acc[sm][sn] = __builtin_amdgcn_wmma_f32_16x16x32_bf16(
                        false, afr[sm], false, bfr[sn], (short)0, acc[sm][sn],
                        false, false);
        }
        __syncthreads();   // all waves done reading buffer cb
    }

    // ---- epilogue ----
    int mb = (lane >> 4) * 8;
#pragma unroll
    for (int sm = 0; sm < 2; sm++) {
#pragma unroll
        for (int sn = 0; sn < 4; sn++) {
            int col = n0 + wn * 64 + sn * 16 + lm;
            float b = bias[(size_t)e * HID + col];
#pragma unroll
            for (int r = 0; r < 8; r++) {
                int row = wm * 32 + sm * 16 + mb + r;
                int p = rowPair[row];
                if (p >= 0) {
                    if (!IS2) {
                        float v = fmaxf(acc[sm][sn][r] + b, 0.0f);
                        hbuf[(size_t)p * HID + col] = (__bf16)v;
                    } else {
                        float v = (acc[sm][sn][r] + b) * pprob[p];
                        unsafeAtomicAdd(&out[(size_t)(p >> 1) * HID + col], v);
                    }
                }
            }
        }
    }
}

// ---------------------------------------------------------------------------
extern "C" void kernel_launch(void* const* d_in, const int* in_sizes, int n_in,
                              void* d_out, int out_size, void* d_ws, size_t ws_size,
                              hipStream_t stream) {
    const float* x  = (const float*)d_in[0];
    const float* Wg = (const float*)d_in[1];
    const float* bg = (const float*)d_in[2];
    const float* W1 = (const float*)d_in[3];
    const float* b1 = (const float*)d_in[4];
    const float* W2 = (const float*)d_in[5];
    const float* b2 = (const float*)d_in[6];

    float* out       = (float*)d_out;                     // [N, H]
    float* probs_out = out + (size_t)NTOK * HID;          // [N, E]
    float* mask_out  = probs_out + (size_t)NTOK * NEXP;   // [N, E]

    // workspace layout (~177 MB total)
    char* w = (char*)d_ws;
    size_t off = 0;
    int* counts = (int*)(w + off);  off += 64;
    int* pairs  = (int*)(w + off);  off += (size_t)NEXP * CAP * 4;   // 512 KB
    float* pprob = (float*)(w + off); off += (size_t)NPAIR * 4;      // 64 KB
    off = (off + 255) & ~(size_t)255;
    __bf16* xb   = (__bf16*)(w + off); off += (size_t)NTOK * DIN * 2;       // 16 MB
    __bf16* w1t  = (__bf16*)(w + off); off += (size_t)NEXP * DIN * HID * 2; // 32 MB
    __bf16* w2t  = (__bf16*)(w + off); off += (size_t)NEXP * HID * HID * 2; // 64 MB
    __bf16* hbuf = (__bf16*)(w + off); off += (size_t)NPAIR * HID * 2;      // 64 MB

    moe_init_kernel<<<2048, 256, 0, stream>>>(out, counts);
    moe_router_kernel<<<NTOK / 8, 256, 0, stream>>>(x, Wg, bg, probs_out, mask_out,
                                                    counts, pairs, pprob);
    moe_cvtx_kernel<<<2048, 256, 0, stream>>>(x, xb);
    // W1 [E][D][H] -> w1t [E][H][D];  W2 [E][H][H] -> w2t [E][H][H] (transposed)
    moe_transpose_kernel<<<dim3(HID / 32, DIN / 32, NEXP), 256, 0, stream>>>(W1, w1t, DIN, HID);
    moe_transpose_kernel<<<dim3(HID / 32, HID / 32, NEXP), 256, 0, stream>>>(W2, w2t, HID, HID);

    dim3 grid(NEXP * TILES_M, HID / BN);
    moe_ffn_kernel<DIN, false><<<grid, 256, 0, stream>>>(xb, w1t, b1, counts, pairs,
                                                         pprob, hbuf, out);
    moe_ffn_kernel<HID, true><<<grid, 256, 0, stream>>>(hbuf, w2t, b2, counts, pairs,
                                                        pprob, hbuf, out);
}
